// NeuralODE_36507222016613
// MI455X (gfx1250) — matrix operand order repaired and checked by reading, not compiled, not run on hardware
//
#include <hip/hip_runtime.h>

typedef __attribute__((ext_vector_type(16))) _Float16 v16h;
typedef __attribute__((ext_vector_type(8)))  _Float16 v8h;
typedef __attribute__((ext_vector_type(8)))  float    v8f;

#define THREADS 256
#define LDA   136   // activation row stride (halves): 128 + 8 pad, 272B = 68 banks -> conflict-free halves
#define LDW   136   // weight row stride (halves) for 128-wide K
#define LDWIH 40    // wih row stride (halves) for 32-wide K (80B = 20 banks/lane, 16 distinct banks)

// ---- problem dims ----
#define NB   512
#define NL   64
#define NT   64
#define NDAT 32

// ---- persistent LDS layout (bytes) during ODE loop ----
#define OFF_VFW0   0        // 128 rows * LDW f16 = 34816
#define OFF_VFW1   34816
#define OFF_VFW2   69632
#define OFF_VFW3   104448
#define OFF_VFB    139264   // 4*128 f32 = 2048
#define OFF_O2W0   141312   // 34816
#define OFF_O2W1   176128   // 34816
#define OFF_O2W2   210944   // 32 rows * LDW f16 = 8704
#define OFF_O2B    219648   // 288 f32 = 1152
#define OFF_ACTA   220800   // 16*LDA f16 = 4352
#define OFF_ACTB   225152   // 4352
#define SMEM_BYTES 229504

// ---- GRU-phase overlay (sequential reuse of [0, OFF_ACTA)) ----
#define OFF_WIH    0        // 384*LDWIH f16 = 30720
#define OFF_WHH    30720    // 384*LDW f16 = 104448 -> 135168
#define OFF_GB     135168   // 384 f32
#define OFF_GBN    136704   // 128 f32
#define OFF_XRAW0  137216   // 16*32 f32 staging (async) = 2048
#define OFF_XRAW1  139264   // 2048
// ---- h2o-phase overlay ----
#define OFF_HW0    0
#define OFF_HW1    34816
#define OFF_HW2    69632
#define OFF_HB     104448   // 384 f32

// Tsit5 tableau
__device__ __constant__ float cA21 = 0.161f;
__device__ __constant__ float cA31 = -0.008480655492356989f, cA32 = 0.335480655492357f;
__device__ __constant__ float cA41 = 2.8971530571054935f, cA42 = -6.359448489975075f, cA43 = 4.3622954328695815f;
__device__ __constant__ float cA51 = 5.325864828439257f, cA52 = -11.748883564062828f,
                             cA53 = 7.4955393428898365f, cA54 = -0.09249506636175525f;
__device__ __constant__ float cA61 = 5.86145544294642f, cA62 = -12.92096931784711f, cA63 = 8.159367898576159f,
                             cA64 = -0.071584973281401f, cA65 = -0.028269050394068383f;
__device__ __constant__ float cB1 = 0.09646076681806523f, cB2 = 0.01f, cB3 = 0.4798896504144996f,
                             cB4 = 1.379008574103742f, cB5 = -3.290069515436081f, cB6 = 2.324710524099774f;

__device__ __forceinline__ v8f wmma16(v16h a, v16h b, v8f c) {
  return __builtin_amdgcn_wmma_f32_16x16x32_f16(false, a, false, b, (short)0, c, false, false);
}

// ---- fast activations: exploit gfx1250 TANH / RCP trans ops ----
__device__ __forceinline__ float fast_tanh(float x) {
#if __has_builtin(__builtin_amdgcn_tanhf)
  return __builtin_amdgcn_tanhf(x);
#else
  x = fminf(15.0f, fmaxf(-15.0f, x));
  const float e = __expf(2.0f * x);
  return (e - 1.0f) * __builtin_amdgcn_rcpf(e + 1.0f);
#endif
}
__device__ __forceinline__ float sigm(float x) {
  return __builtin_amdgcn_rcpf(1.0f + __expf(-x));
}

// ---- async global -> LDS (CDNA5 path, ASYNCcnt) ----
__device__ __forceinline__ void async_load_b128(unsigned lds_byte_addr, const float* sbase, unsigned voff) {
  asm volatile("global_load_async_to_lds_b128 %0, %1, %2"
               :: "v"(lds_byte_addr), "v"(voff), "s"(sbase)
               : "memory");
}
__device__ __forceinline__ void wait_asynccnt0() {
  asm volatile("s_wait_asynccnt 0x0" ::: "memory");
}

// A fragment (16x32 tile of a 16-row activation matrix in LDS, f16).
__device__ __forceinline__ v16h load_a_frag(const _Float16* X, int ldx, int kb) {
  const int lane = threadIdx.x & 31;
  const int row  = lane & 15;
  const int ko   = (lane & 16) ? 8 : 0;
  const _Float16* p = X + row * ldx + kb + ko;
  union { v16h v; v8h h[2]; } u;
  u.h[0] = *(const v8h*)(p);
  u.h[1] = *(const v8h*)(p + 16);
  return u.v;
}

// B fragment for y = x @ W^T : B[k][n] = W[n][k]; W row-major, padded row stride ldw (f16 in LDS).
__device__ __forceinline__ v16h load_b_frag(const _Float16* W, int ldw, int n0col, int kb) {
  const int lane = threadIdx.x & 31;
  const int col  = n0col + (lane & 15);
  const int ko   = (lane & 16) ? 8 : 0;
  const _Float16* p = W + col * ldw + kb + ko;
  union { v16h v; v8h h[2]; } u;
  u.h[0] = *(const v8h*)(p);
  u.h[1] = *(const v8h*)(p + 16);
  return u.v;
}

__device__ __forceinline__ v8f zero8() {
  v8f c;
#pragma unroll
  for (int i = 0; i < 8; ++i) c[i] = 0.0f;
  return c;
}

// 16x16 output tile of [16,128] @ W^T, K=128
__device__ __forceinline__ v8f mm128(const _Float16* X, const _Float16* W, int n0col) {
  v8f c = zero8();
#pragma unroll
  for (int kb = 0; kb < 128; kb += 32)
    c = wmma16(load_a_frag(X, LDA, kb), load_b_frag(W, LDW, n0col, kb), c);
  return c;
}

__device__ __forceinline__ void store_act(_Float16* dst, int n0, v8f c, float bias, bool dotanh) {
  const int lane  = threadIdx.x & 31;
  const int n     = n0 + (lane & 15);
  const int mbase = (lane & 16) ? 8 : 0;
#pragma unroll
  for (int r = 0; r < 8; ++r) {
    float v = c[r] + bias;
    if (dotanh) v = fast_tanh(v);
    dst[(mbase + r) * LDA + n] = (_Float16)v;
  }
}

// global f32 -> padded LDS f16 weight matrix (cols = 1<<cshift logical columns per row)
__device__ __forceinline__ void g2l_w16(_Float16* dst, const float* src, int n, int cshift, int ldw, int tid) {
  const int cmask = (1 << cshift) - 1;
  for (int i = tid; i < n; i += THREADS)
    dst[(i >> cshift) * ldw + (i & cmask)] = (_Float16)src[i];
}
__device__ __forceinline__ void g2l_f32(float* dst, const float* src, int n, int tid) {
  for (int i = tid; i < n; i += THREADS) dst[i] = src[i];
}

// one vector-field MLP evaluation; stage input already in actA (f16), barrier done.
__device__ __forceinline__ v8f vf_eval(_Float16* actA, _Float16* actB,
                                       const _Float16* W0, const _Float16* W1,
                                       const _Float16* W2, const _Float16* W3,
                                       const float* vb, int n0, int nn) {
  v8f c = mm128(actA, W0, n0); store_act(actB, n0, c, vb[nn],       true); __syncthreads();
  c     = mm128(actB, W1, n0); store_act(actA, n0, c, vb[128 + nn], true); __syncthreads();
  c     = mm128(actA, W2, n0); store_act(actB, n0, c, vb[256 + nn], true); __syncthreads();
  c     = mm128(actB, W3, n0);
  const float b3 = vb[384 + nn];
#pragma unroll
  for (int r = 0; r < 8; ++r) c[r] += b3;
  return c;
}

// ode_to_data (3 linear layers) applied to y (registers) -> d_out at save index t
__device__ __forceinline__ void o2d_save(v8f yreg, _Float16* actA, _Float16* actB,
                                         const _Float16* W0, const _Float16* W1, const _Float16* W2,
                                         const float* ob, float* out,
                                         int mb, int t, int n0, int nn, int wave) {
  store_act(actA, n0, yreg, 0.0f, false); __syncthreads();
  v8f c = mm128(actA, W0, n0); store_act(actB, n0, c, ob[nn],       false); __syncthreads();
  c     = mm128(actB, W1, n0); store_act(actA, n0, c, ob[128 + nn], false); __syncthreads();
  if (wave < 2) {  // output width 32: waves 0,1 only (uniform per wave -> EXEC all-ones for WMMA)
    c = mm128(actA, W2, n0);
    const float b2    = ob[256 + nn];
    const int   mbase = (threadIdx.x & 16) ? 8 : 0;
#pragma unroll
    for (int r = 0; r < 8; ++r)
      out[((size_t)(mb + mbase + r) * NT + t) * NDAT + nn] = c[r] + b2;
  }
  __syncthreads();
}

__global__ __launch_bounds__(THREADS) void node_fused(
    const float* __restrict__ ts, const float* __restrict__ yi,
    const float* __restrict__ wih, const float* __restrict__ whh,
    const float* __restrict__ gbias, const float* __restrict__ gbias_n,
    const float* __restrict__ h2oW0, const float* __restrict__ h2ob0,
    const float* __restrict__ h2oW1, const float* __restrict__ h2ob1,
    const float* __restrict__ h2oW2, const float* __restrict__ h2ob2,
    const float* __restrict__ vfW0, const float* __restrict__ vfb0,
    const float* __restrict__ vfW1, const float* __restrict__ vfb1,
    const float* __restrict__ vfW2, const float* __restrict__ vfb2,
    const float* __restrict__ vfW3, const float* __restrict__ vfb3,
    const float* __restrict__ o2W0, const float* __restrict__ o2b0,
    const float* __restrict__ o2W1, const float* __restrict__ o2b1,
    const float* __restrict__ o2W2, const float* __restrict__ o2b2,
    float* __restrict__ out) {
  extern __shared__ char smem[];
  const int tid  = threadIdx.x;
  const int wave = tid >> 5;
  const int lane = tid & 31;
  const int n0   = wave * 16;
  const int nn   = n0 + (lane & 15);
  const int mb   = blockIdx.x * 16;

  _Float16* actA = (_Float16*)(smem + OFF_ACTA);
  _Float16* actB = (_Float16*)(smem + OFF_ACTB);

  // Warm L2 (global_prefetch_b8) for the weights used by later phases while GRU runs.
  if (tid < 64) {
    __builtin_prefetch(vfW0 + tid * 256, 0, 1);
    __builtin_prefetch(vfW1 + tid * 256, 0, 1);
    __builtin_prefetch(vfW2 + tid * 256, 0, 1);
    __builtin_prefetch(vfW3 + tid * 256, 0, 1);
    __builtin_prefetch(o2W0 + tid * 256, 0, 1);
    __builtin_prefetch(o2W1 + tid * 256, 0, 1);
  }

  // ================= Phase 1: GRU encoder (reversed sequence) =================
  {
    _Float16* wih16 = (_Float16*)(smem + OFF_WIH);
    _Float16* whh16 = (_Float16*)(smem + OFF_WHH);
    float*    gb    = (float*)(smem + OFF_GB);
    float*    gbn   = (float*)(smem + OFF_GBN);
    _Float16* h16   = actA;                    // 16 x LDA
    _Float16* x16   = actB;                    // 16 x 32

    g2l_w16(wih16, wih, 384 * 32, 5, LDWIH, tid);
    g2l_w16(whh16, whh, 384 * 128, 7, LDW, tid);
    g2l_f32(gb, gbias, 384, tid);
    g2l_f32(gbn, gbias_n, 128, tid);
    for (int i = tid; i < 16 * LDA; i += THREADS) h16[i] = (_Float16)0.0f;

    // kick off async prefetch of x for t=0 (waves 0..3: 128 lanes x 16B = 2KB tile)
    if (tid < 128) {
      const unsigned goff =
          (unsigned)(((((mb + (tid >> 3)) * NL + (NL - 1)) * NDAT) + ((tid & 7) * 4)) * 4);
      async_load_b128((unsigned)OFF_XRAW0 + (unsigned)tid * 16u, yi, goff);
    }
    __syncthreads();

    const float bR = gb[nn], bZ = gb[128 + nn], bN = gb[256 + nn], bNn = gbn[nn];
    v8f hreg = zero8();
    const int mbase = (lane & 16) ? 8 : 0;

    for (int t = 0; t < NL; ++t) {
      wait_asynccnt0();      // issuing waves drain their ASYNCcnt (no-op for others)
      __syncthreads();       // publish async LDS writes to all waves

      // convert staged f32 tile -> f16 x16, and prefetch the next tile
      const float* xr = (const float*)(smem + ((t & 1) ? OFF_XRAW1 : OFF_XRAW0));
      for (int i = tid; i < 16 * 32; i += THREADS) x16[i] = (_Float16)xr[i];
      if (t + 1 < NL && tid < 128) {
        const int st2 = NL - 2 - t;
        const unsigned goff =
            (unsigned)(((((mb + (tid >> 3)) * NL + st2) * NDAT) + ((tid & 7) * 4)) * 4);
        const unsigned ldst = (unsigned)((t & 1) ? OFF_XRAW0 : OFF_XRAW1) + (unsigned)tid * 16u;
        async_load_b128(ldst, yi, goff);
      }
      __syncthreads();

      const v16h ax = load_a_frag(x16, 32, 0);
      v16h ah[4];
#pragma unroll
      for (int k = 0; k < 4; ++k) ah[k] = load_a_frag(h16, LDA, k * 32);

      v8f accR = zero8(), accZ = zero8(), accNi = zero8(), accNh = zero8();
      accR  = wmma16(ax, load_b_frag(wih16, LDWIH, n0, 0),       accR);
      accZ  = wmma16(ax, load_b_frag(wih16, LDWIH, 128 + n0, 0), accZ);
      accNi = wmma16(ax, load_b_frag(wih16, LDWIH, 256 + n0, 0), accNi);
#pragma unroll
      for (int k = 0; k < 4; ++k) {
        accR  = wmma16(ah[k], load_b_frag(whh16, LDW, n0,       k * 32), accR);
        accZ  = wmma16(ah[k], load_b_frag(whh16, LDW, 128 + n0, k * 32), accZ);
        accNh = wmma16(ah[k], load_b_frag(whh16, LDW, 256 + n0, k * 32), accNh);
      }
      __syncthreads();  // all reads of h16/x16 complete before rewrite

#pragma unroll
      for (int r = 0; r < 8; ++r) {
        const float rg = sigm(accR[r] + bR);
        const float zg = sigm(accZ[r] + bZ);
        const float ng = fast_tanh(accNi[r] + bN + bNn + rg * accNh[r]);
        hreg[r] = ng + zg * (hreg[r] - ng);
      }
#pragma unroll
      for (int r = 0; r < 8; ++r) h16[(mbase + r) * LDA + nn] = (_Float16)hreg[r];
      __syncthreads();
    }
  }

  // ================= Phase 2: hidden_to_ode MLP (tanh,tanh,linear) =================
  v8f yreg;
  {
    _Float16* hW0 = (_Float16*)(smem + OFF_HW0);
    _Float16* hW1 = (_Float16*)(smem + OFF_HW1);
    _Float16* hW2 = (_Float16*)(smem + OFF_HW2);
    float*    hb  = (float*)(smem + OFF_HB);
    g2l_w16(hW0, h2oW0, 128 * 128, 7, LDW, tid);
    g2l_w16(hW1, h2oW1, 128 * 128, 7, LDW, tid);
    g2l_w16(hW2, h2oW2, 128 * 128, 7, LDW, tid);
    g2l_f32(hb, h2ob0, 128, tid);
    g2l_f32(hb + 128, h2ob1, 128, tid);
    g2l_f32(hb + 256, h2ob2, 128, tid);
    __syncthreads();

    v8f c = mm128(actA, hW0, n0); store_act(actB, n0, c, hb[nn],       true); __syncthreads();
    c     = mm128(actB, hW1, n0); store_act(actA, n0, c, hb[128 + nn], true); __syncthreads();
    c     = mm128(actA, hW2, n0);
    const float b2 = hb[256 + nn];
#pragma unroll
    for (int r = 0; r < 8; ++r) yreg[r] = c[r] + b2;
    __syncthreads();  // h2o weights consumed; safe to overlay
  }

  // ================= Phase 3: load vf + o2d weights (resident for ODE loop) =================
  _Float16* vW0 = (_Float16*)(smem + OFF_VFW0);
  _Float16* vW1 = (_Float16*)(smem + OFF_VFW1);
  _Float16* vW2 = (_Float16*)(smem + OFF_VFW2);
  _Float16* vW3 = (_Float16*)(smem + OFF_VFW3);
  float*    vb  = (float*)(smem + OFF_VFB);
  _Float16* oW0 = (_Float16*)(smem + OFF_O2W0);
  _Float16* oW1 = (_Float16*)(smem + OFF_O2W1);
  _Float16* oW2 = (_Float16*)(smem + OFF_O2W2);
  float*    ob  = (float*)(smem + OFF_O2B);
  g2l_w16(vW0, vfW0, 128 * 128, 7, LDW, tid);
  g2l_w16(vW1, vfW1, 128 * 128, 7, LDW, tid);
  g2l_w16(vW2, vfW2, 128 * 128, 7, LDW, tid);
  g2l_w16(vW3, vfW3, 128 * 128, 7, LDW, tid);
  g2l_f32(vb, vfb0, 128, tid);
  g2l_f32(vb + 128, vfb1, 128, tid);
  g2l_f32(vb + 256, vfb2, 128, tid);
  g2l_f32(vb + 384, vfb3, 128, tid);
  g2l_w16(oW0, o2W0, 128 * 128, 7, LDW, tid);
  g2l_w16(oW1, o2W1, 128 * 128, 7, LDW, tid);
  g2l_w16(oW2, o2W2, 32 * 128, 7, LDW, tid);
  g2l_f32(ob, o2b0, 128, tid);
  g2l_f32(ob + 128, o2b1, 128, tid);
  g2l_f32(ob + 256, o2b2, 32, tid);
  __syncthreads();

  // ================= Phase 4: Tsit5 integration with fused o2d saves =================
  o2d_save(yreg, actA, actB, oW0, oW1, oW2, ob, out, mb, 0, n0, nn, wave);

  for (int ti = 0; ti < NT - 1; ++ti) {
    const float hh = (ts[ti + 1] - ts[ti]) * 0.5f;  // dt / SUBSTEPS
    for (int ss = 0; ss < 2; ++ss) {
      v8f k1, k2, k3, k4, k5, k6, yt;

      store_act(actA, n0, yreg, 0.0f, false); __syncthreads();
      k1 = vf_eval(actA, actB, vW0, vW1, vW2, vW3, vb, n0, nn);

      yt = yreg + (hh * cA21) * k1;
      store_act(actA, n0, yt, 0.0f, false); __syncthreads();
      k2 = vf_eval(actA, actB, vW0, vW1, vW2, vW3, vb, n0, nn);

      yt = yreg + hh * (cA31 * k1 + cA32 * k2);
      store_act(actA, n0, yt, 0.0f, false); __syncthreads();
      k3 = vf_eval(actA, actB, vW0, vW1, vW2, vW3, vb, n0, nn);

      yt = yreg + hh * (cA41 * k1 + cA42 * k2 + cA43 * k3);
      store_act(actA, n0, yt, 0.0f, false); __syncthreads();
      k4 = vf_eval(actA, actB, vW0, vW1, vW2, vW3, vb, n0, nn);

      yt = yreg + hh * (cA51 * k1 + cA52 * k2 + cA53 * k3 + cA54 * k4);
      store_act(actA, n0, yt, 0.0f, false); __syncthreads();
      k5 = vf_eval(actA, actB, vW0, vW1, vW2, vW3, vb, n0, nn);

      yt = yreg + hh * (cA61 * k1 + cA62 * k2 + cA63 * k3 + cA64 * k4 + cA65 * k5);
      store_act(actA, n0, yt, 0.0f, false); __syncthreads();
      k6 = vf_eval(actA, actB, vW0, vW1, vW2, vW3, vb, n0, nn);

      yreg = yreg + hh * (cB1 * k1 + cB2 * k2 + cB3 * k3 + cB4 * k4 + cB5 * k5 + cB6 * k6);
    }
    o2d_save(yreg, actA, actB, oW0, oW1, oW2, ob, out, mb, ti + 1, n0, nn, wave);
  }
}

extern "C" void kernel_launch(void* const* d_in, const int* in_sizes, int n_in,
                              void* d_out, int out_size, void* d_ws, size_t ws_size,
                              hipStream_t stream) {
  (void)in_sizes; (void)n_in; (void)out_size; (void)d_ws; (void)ws_size;
  const float* ts    = (const float*)d_in[0];
  const float* yi    = (const float*)d_in[1];
  const float* wih   = (const float*)d_in[2];
  const float* whh   = (const float*)d_in[3];
  const float* gbias = (const float*)d_in[4];
  const float* gbn   = (const float*)d_in[5];
  const float* h2oW0 = (const float*)d_in[6];
  const float* h2ob0 = (const float*)d_in[7];
  const float* h2oW1 = (const float*)d_in[8];
  const float* h2ob1 = (const float*)d_in[9];
  const float* h2oW2 = (const float*)d_in[10];
  const float* h2ob2 = (const float*)d_in[11];
  const float* vfW0  = (const float*)d_in[12];
  const float* vfb0  = (const float*)d_in[13];
  const float* vfW1  = (const float*)d_in[14];
  const float* vfb1  = (const float*)d_in[15];
  const float* vfW2  = (const float*)d_in[16];
  const float* vfb2  = (const float*)d_in[17];
  const float* vfW3  = (const float*)d_in[18];
  const float* vfb3  = (const float*)d_in[19];
  const float* o2W0  = (const float*)d_in[20];
  const float* o2b0  = (const float*)d_in[21];
  const float* o2W1  = (const float*)d_in[22];
  const float* o2b1  = (const float*)d_in[23];
  const float* o2W2  = (const float*)d_in[24];
  const float* o2b2  = (const float*)d_in[25];
  float* out = (float*)d_out;

  (void)hipFuncSetAttribute(reinterpret_cast<const void*>(node_fused),
                            hipFuncAttributeMaxDynamicSharedMemorySize, SMEM_BYTES);

  node_fused<<<NB / 16, THREADS, SMEM_BYTES, stream>>>(
      ts, yi, wih, whh, gbias, gbn,
      h2oW0, h2ob0, h2oW1, h2ob1, h2oW2, h2ob2,
      vfW0, vfb0, vfW1, vfb1, vfW2, vfb2, vfW3, vfb3,
      o2W0, o2b0, o2W1, o2b1, o2W2, o2b2, out);
}
